// GNN_old_14465449853060
// MI455X (gfx1250) — matrix operand order repaired and checked by reading, not compiled, not run on hardware
//
#include <hip/hip_runtime.h>
#include <hip/hip_bf16.h>

typedef __attribute__((ext_vector_type(16))) __bf16 v16bf;
typedef __attribute__((ext_vector_type(8)))  float  v8f;

#define N_NODES 50000
#define N_EDGES 800000
#define DFEAT   64

// ---------------------------------------------------------------------------
// Zero a float buffer (workspace is poisoned by the harness).
// ---------------------------------------------------------------------------
__global__ void zero_f32(float* __restrict__ p, int n) {
    int i = blockIdx.x * blockDim.x + threadIdx.x;
    int stride = gridDim.x * blockDim.x;
    for (; i < n; i += stride) p[i] = 0.0f;
}

// ---------------------------------------------------------------------------
// Edge scatter: one wave32 per edge. Each lane adds 2 consecutive features of
// the source row into agg[dst] with global_atomic_add_f32 (the 12.8 MB target
// is L2-resident on the 192 MB L2, so these are L2 atomics, not HBM traffic).
// Lane 0 accumulates the in-degree (layer 1 only; graph identical in layer 2).
// ---------------------------------------------------------------------------
__global__ void scatter_edges(const float* __restrict__ feat,
                              const int*   __restrict__ ei,
                              float* __restrict__ agg,
                              float* __restrict__ cnt,
                              int nEdges, int addCount) {
    long long t = (long long)blockIdx.x * blockDim.x + threadIdx.x;
    int e    = (int)(t >> 5);
    int lane = (int)(t & 31);
    if (e >= nEdges) return;
    int src = ei[e];
    int dst = ei[nEdges + e];
    const float2 v = *(const float2*)(feat + (size_t)src * DFEAT + lane * 2);
    float* dp = agg + (size_t)dst * DFEAT + lane * 2;
    atomicAdd(dp + 0, v.x);
    atomicAdd(dp + 1, v.y);
    if (addCount && lane == 0) atomicAdd(cnt + dst, 1.0f);
}

// ---------------------------------------------------------------------------
// Build a 16x32 bf16 A-fragment for v_wmma_f32_16x16x32_bf16 from one fp32
// row. ISA 16-bit A layout: lane L holds row M=L&15; lane-half selects K+8;
// element e -> k = (e&7) + ((L&16)>>1) + ((e&8)<<1).  Per lane that is two
// contiguous 8-float runs at p[off+kb..] and p[off+kb+16..] -> 4x b128 loads
// (off/kb are compile-time or loop-invariant, so they fold into IOFFSET).
// The mean-normalization scale is folded in before the bf16 round.
// ---------------------------------------------------------------------------
__device__ __forceinline__ v16bf make_a_frag(const float* __restrict__ p,
                                             int off, float sc) {
    float4 f0 = *(const float4*)(p + off);
    float4 f1 = *(const float4*)(p + off + 4);
    float4 f2 = *(const float4*)(p + off + 16);
    float4 f3 = *(const float4*)(p + off + 20);
    v16bf a;
    a[0]  = (__bf16)(f0.x * sc); a[1]  = (__bf16)(f0.y * sc);
    a[2]  = (__bf16)(f0.z * sc); a[3]  = (__bf16)(f0.w * sc);
    a[4]  = (__bf16)(f1.x * sc); a[5]  = (__bf16)(f1.y * sc);
    a[6]  = (__bf16)(f1.z * sc); a[7]  = (__bf16)(f1.w * sc);
    a[8]  = (__bf16)(f2.x * sc); a[9]  = (__bf16)(f2.y * sc);
    a[10] = (__bf16)(f2.z * sc); a[11] = (__bf16)(f2.w * sc);
    a[12] = (__bf16)(f3.x * sc); a[13] = (__bf16)(f3.y * sc);
    a[14] = (__bf16)(f3.z * sc); a[15] = (__bf16)(f3.w * sc);
    return a;
}

// ---------------------------------------------------------------------------
// Fused SAGE layer: hOut = relu( (agg/cnt) @ Wl + b + hIn @ Wr ).
// 256 threads = 8 wave32; one wave per 16-node tile (50000 = 3125*16 exact).
// Weights converted fp32->bf16 into LDS once per block; each wave then keeps
// all 16 B-fragments (2 weights x 4 N-tiles x 2 K-steps) in registers laid
// out per the ISA B layout: lane L holds col N=L&15, lane-half selects K+8,
// element e walks K.
// ---------------------------------------------------------------------------
__global__ __launch_bounds__(256)
void sage_layer(const float* __restrict__ agg,
                const float* __restrict__ cnt,
                const float* __restrict__ hIn,
                const float* __restrict__ Wl,    // [64][64] k-major
                const float* __restrict__ bias,  // [64]
                const float* __restrict__ Wr,    // [64][64]
                float* __restrict__ hOut,
                int nTiles) {
    __shared__ __bf16 sWl[DFEAT * DFEAT];
    __shared__ __bf16 sWr[DFEAT * DFEAT];
    for (int i = threadIdx.x; i < DFEAT * DFEAT; i += blockDim.x) {
        sWl[i] = (__bf16)Wl[i];
        sWr[i] = (__bf16)Wr[i];
    }
    __syncthreads();

    const int lane = threadIdx.x & 31;
    const int wave = threadIdx.x >> 5;
    const int m    = lane & 15;          // A row / B col / C col within tile
    const int hi   = (lane >> 4) & 1;    // lane half
    const int kbA  = hi * 8;             // A fragment K base
    const int kbB  = hi * 8;             // B fragment K base

    // Preload B fragments (weights are reused across every node tile).
    v16bf Bl[2][4], Br[2][4];
#pragma unroll
    for (int s = 0; s < 2; ++s) {
#pragma unroll
        for (int j = 0; j < 4; ++j) {
            const int k0 = s * 32 + kbB;
            const int n  = j * 16 + m;
#pragma unroll
            for (int e = 0; e < 16; ++e) {
                Bl[s][j][e] = sWl[(k0 + (e & 7) + ((e & 8) << 1)) * DFEAT + n];
                Br[s][j][e] = sWr[(k0 + (e & 7) + ((e & 8) << 1)) * DFEAT + n];
            }
        }
    }

    float bv[4];
#pragma unroll
    for (int j = 0; j < 4; ++j) bv[j] = bias[j * 16 + m];

    for (int tile = blockIdx.x * 8 + wave; tile < nTiles; tile += gridDim.x * 8) {
        const int nodeBase = tile * 16;
        const int rowA = nodeBase + m;   // row this lane loads for A fragments
        const float c  = cnt[rowA];
        const float sc = 1.0f / fmaxf(c, 1.0f);

        // Loop-invariant base pointers: everything below addresses via
        // compile-time IOFFSETs off these three bases.
        const float* pa   = agg  + (size_t)rowA * DFEAT + kbA;
        const float* ph   = hIn  + (size_t)rowA * DFEAT + kbA;
        float*       pout = hOut + (size_t)(nodeBase + hi * 8) * DFEAT + m;

        v8f acc[4];
#pragma unroll
        for (int j = 0; j < 4; ++j) {
            v8f a = {bv[j], bv[j], bv[j], bv[j], bv[j], bv[j], bv[j], bv[j]};
            acc[j] = a;
        }

#pragma unroll
        for (int s = 0; s < 2; ++s) {
            v16bf Aa = make_a_frag(pa, s * 32, sc);    // mean-aggregated term
            v16bf As = make_a_frag(ph, s * 32, 1.0f);  // root/self term
#pragma unroll
            for (int j = 0; j < 4; ++j) {
                acc[j] = __builtin_amdgcn_wmma_f32_16x16x32_bf16(
                    false, Aa, false, Bl[s][j], (short)0, acc[j], false, false);
                acc[j] = __builtin_amdgcn_wmma_f32_16x16x32_bf16(
                    false, As, false, Br[s][j], (short)0, acc[j], false, false);
            }
        }

        // C/D layout: lane holds col n = m, VGPR r holds row r + 8*hi.
        // All 32 stores are constant offsets (r*256 + j*64 bytes) off pout.
#pragma unroll
        for (int j = 0; j < 4; ++j) {
#pragma unroll
            for (int r = 0; r < 8; ++r) {
                pout[r * DFEAT + j * 16] = fmaxf(acc[j][r], 0.0f);
            }
        }
    }
}

// ---------------------------------------------------------------------------
// Final head: out = h2 @ Wlin + blin  (64 -> 2). One thread per node.
// ---------------------------------------------------------------------------
__global__ void final_linear(const float* __restrict__ h,
                             const float* __restrict__ W,   // [64][2]
                             const float* __restrict__ b,   // [2]
                             float* __restrict__ out, int n) {
    int i = blockIdx.x * blockDim.x + threadIdx.x;
    if (i >= n) return;
    const float* row = h + (size_t)i * DFEAT;
    float a0 = b[0], a1 = b[1];
#pragma unroll
    for (int k = 0; k < DFEAT; k += 4) {
        float4 v = *(const float4*)(row + k);
        a0 += v.x * W[(k + 0) * 2 + 0] + v.y * W[(k + 1) * 2 + 0] +
              v.z * W[(k + 2) * 2 + 0] + v.w * W[(k + 3) * 2 + 0];
        a1 += v.x * W[(k + 0) * 2 + 1] + v.y * W[(k + 1) * 2 + 1] +
              v.z * W[(k + 2) * 2 + 1] + v.w * W[(k + 3) * 2 + 1];
    }
    out[(size_t)i * 2 + 0] = a0;
    out[(size_t)i * 2 + 1] = a1;
}

// ---------------------------------------------------------------------------
extern "C" void kernel_launch(void* const* d_in, const int* in_sizes, int n_in,
                              void* d_out, int out_size, void* d_ws, size_t ws_size,
                              hipStream_t stream) {
    const float* x    = (const float*)d_in[0];
    const int*   ei   = (const int*)d_in[1];
    const float* W1l  = (const float*)d_in[2];
    const float* b1   = (const float*)d_in[3];
    const float* W1r  = (const float*)d_in[4];
    const float* W2l  = (const float*)d_in[5];
    const float* b2   = (const float*)d_in[6];
    const float* W2r  = (const float*)d_in[7];
    const float* Wlin = (const float*)d_in[8];
    const float* blin = (const float*)d_in[9];
    float* out = (float*)d_out;

    // Workspace layout: cnt[N] | agg[N*64] | h1[N*64] | h2[N*64]  (~38.6 MB)
    float* cnt = (float*)d_ws;
    float* agg = cnt + N_NODES;
    float* h1  = agg + (size_t)N_NODES * DFEAT;
    float* h2  = h1 + (size_t)N_NODES * DFEAT;

    const int nTiles = N_NODES / 16;                 // 3125, exact
    const int sageBlocks = (nTiles + 7) / 8;         // 8 waves (tiles) / block
    const long long scatterThreads = (long long)N_EDGES * 32;
    const int scatterBlocks = (int)((scatterThreads + 255) / 256);

    // Layer 1
    zero_f32<<<1024, 256, 0, stream>>>(cnt, N_NODES * (DFEAT + 1)); // cnt+agg contiguous
    scatter_edges<<<scatterBlocks, 256, 0, stream>>>(x, ei, agg, cnt, N_EDGES, 1);
    sage_layer<<<sageBlocks, 256, 0, stream>>>(agg, cnt, x, W1l, b1, W1r, h1, nTiles);

    // Layer 2 (reuse agg buffer and degree counts)
    zero_f32<<<1024, 256, 0, stream>>>(agg, N_NODES * DFEAT);
    scatter_edges<<<scatterBlocks, 256, 0, stream>>>(h1, ei, agg, cnt, N_EDGES, 0);
    sage_layer<<<sageBlocks, 256, 0, stream>>>(agg, cnt, h1, W2l, b2, W2r, h2, nTiles);

    // Head
    final_linear<<<(N_NODES + 255) / 256, 256, 0, stream>>>(h2, Wlin, blin, out, N_NODES);
}
